// Seq2Seq_84696755077219
// MI455X (gfx1250) — matrix operand order repaired and checked by reading, not compiled, not run on hardware
//
#include <hip/hip_runtime.h>
#include <hip/hip_bf16.h>

// ---------------- problem constants ----------------
#define V_   32000
#define E_   1024
#define H_   1024
#define L_   400
#define T_   100
#define TPAD 112      // T padded to multiple of 16 for WMMA N-tiles
#define SOS_ 1

typedef __attribute__((ext_vector_type(16))) _Float16 v16h;
typedef __attribute__((ext_vector_type(8)))  float    v8f;

// =====================================================================
// B operands are stored pre-swizzled in WMMA fragment-major layout:
//   fragment index  f = kTile * nTiles + nTile      (kTile = k/32, nTile = n/16)
//   within fragment: lane (0..31), half p (0..15)
//   element (k, n):  k = kTile*32 + (lane>=16)*16 + p,  n = nTile*16 + (lane&15)
// so the GEMM reads one aligned 32-byte v16h per lane per fragment.
// =====================================================================
__device__ __forceinline__ size_t bswz_idx(int k, int n, int nTiles) {
    int kT = k >> 5, kin = k & 31;
    int halfi = kin >> 4, p = kin & 15;
    int nT = n >> 4, l16 = n & 15;
    int lane = halfi * 16 + l16;
    return ((size_t)(kT * nTiles + nT) * 32 + lane) * 16 + p;
}

// =====================================================================
// WMMA GEMM:  Out[n*M + m] = bias[m] + sum_k A[m*lda+k] * B(k,n)
// A: f32 row-major (converted to f16 per-fragment), B: swizzled v16h frags.
// One wave per 16(M) x 32(N) output macro-tile: A fragments are reused
// across two N-tiles (halves f32 A traffic + cvt work per WMMA).
// K looped in chunks of 64 -> 4 interleaved WMMAs per iteration.
// =====================================================================
__global__ __launch_bounds__(32)
void wmma_gemm_kernel(const float* __restrict__ A, int lda,
                      const v16h* __restrict__ Bsw, int nTiles,
                      const float* __restrict__ bias,
                      float* __restrict__ Out,
                      int M, int K, int Nreal)
{
    const int lane  = threadIdx.x & 31;
    const int l16   = lane & 15;
    const int halfi = lane >> 4;          // 0: lanes 0-15, 1: lanes 16-31
    const int m0  = blockIdx.x * 16;
    const int nT0 = blockIdx.y * 2;
    const int nT1 = nT0 + 1;
    const bool has1 = (nT1 < nTiles);

    // C/D layout: VGPR r -> m = m0 + r + 8*halfi, n = nT*16 + l16
    v8f c0, c1;
#pragma unroll
    for (int r = 0; r < 8; ++r) {
        int m = m0 + r + halfi * 8;
        float b = bias ? bias[m] : 0.0f;
        c0[r] = b;
        c1[r] = b;
    }

    const int mA    = m0 + l16;          // A fragment row (same for both halves)
    const int kOffA = halfi * 16;        // A: lane half h covers K+16h..16h+15 contiguously
    const float* Arow = A + (size_t)mA * lda + kOffA;
    const v16h*  B0   = Bsw + (size_t)nT0 * 32 + lane;
    const v16h*  B1   = Bsw + (size_t)(has1 ? nT1 : nT0) * 32 + lane;
    const size_t fstep = (size_t)nTiles * 32;   // fragment stride per K-tile

    for (int kb = 0; kb < K; kb += 64) {
        v16h a0, a1;
        const float* Ap0 = Arow + kb;
        const float* Ap1 = Arow + kb + 32;
#pragma unroll
        for (int p = 0; p < 8; ++p) {
            a0[p]     = (_Float16)Ap0[p];
            a0[p + 8] = (_Float16)Ap0[p + 8];
            a1[p]     = (_Float16)Ap1[p];
            a1[p + 8] = (_Float16)Ap1[p + 8];
        }
        const size_t f0 = (size_t)(kb >> 5) * fstep;
        const size_t f1 = f0 + fstep;
        v16h b00 = B0[f0];
        v16h b10 = B1[f0];
        v16h b01 = B0[f1];
        v16h b11 = B1[f1];
        // interleave the two accumulator chains so the WMMA RAW chains pipeline
        c0 = __builtin_amdgcn_wmma_f32_16x16x32_f16(false, a0, false, b00, (short)0, c0, false, false);
        c1 = __builtin_amdgcn_wmma_f32_16x16x32_f16(false, a0, false, b10, (short)0, c1, false, false);
        c0 = __builtin_amdgcn_wmma_f32_16x16x32_f16(false, a1, false, b01, (short)0, c0, false, false);
        c1 = __builtin_amdgcn_wmma_f32_16x16x32_f16(false, a1, false, b11, (short)0, c1, false, false);
    }

    const int n0 = nT0 * 16 + l16;
    if (n0 < Nreal) {
#pragma unroll
        for (int r = 0; r < 8; ++r) {
            int m = m0 + r + halfi * 8;
            Out[(size_t)n0 * M + m] = c0[r];
        }
    }
    if (has1) {
        const int n1 = nT1 * 16 + l16;
        if (n1 < Nreal) {
#pragma unroll
            for (int r = 0; r < 8; ++r) {
                int m = m0 + r + halfi * 8;
                Out[(size_t)n1 * M + m] = c1[r];
            }
        }
    }
}

// =====================================================================
// Row-wise matvec with LDS reduction: y[row] = act(dot(W[row,:],x) + bias + addv)
// float4 vector loads (all lengths here are multiples of 4, rows 16B aligned).
// =====================================================================
__global__ __launch_bounds__(256)
void matvec_kernel(const float* __restrict__ W, int ldw,
                   const float* __restrict__ x, int n,
                   const float* __restrict__ bias,
                   const float* __restrict__ addv,
                   float* __restrict__ y, int relu_flag)
{
    __shared__ float red[256];
    const int row = blockIdx.x;
    const float4* w4 = (const float4*)(W + (size_t)row * ldw);
    const float4* x4 = (const float4*)x;
    const int n4 = n >> 2;
    float s = 0.0f;
    for (int i = threadIdx.x; i < n4; i += 256) {
        float4 a = w4[i], b = x4[i];
        s += a.x * b.x + a.y * b.y + a.z * b.z + a.w * b.w;
    }
    red[threadIdx.x] = s;
    __syncthreads();
    for (int w = 128; w > 0; w >>= 1) {
        if (threadIdx.x < w) red[threadIdx.x] += red[threadIdx.x + w];
        __syncthreads();
    }
    if (threadIdx.x == 0) {
        float v = red[0];
        if (bias) v += bias[row];
        if (addv) v += addv[row];
        if (relu_flag) v = fmaxf(v, 0.0f);
        y[row] = v;
    }
}

// ctx[j] = sum_l aw[l] * encOuts[l*H + j]
__global__ __launch_bounds__(256)
void ctx_kernel(const float* __restrict__ encOuts,
                const float* __restrict__ aw,
                float* __restrict__ ctx)
{
    int j = blockIdx.x * 256 + threadIdx.x;
    if (j >= H_) return;
    float s = 0.0f;
    for (int l = 0; l < L_; ++l) s += aw[l] * encOuts[(size_t)l * H_ + j];
    ctx[j] = s;
}

// PyTorch GRU cell gate math; updates h in place; optionally saves h to an
// f32 row (encoder outputs) and/or the swizzled f16 Hb column (decoder H2).
__global__ __launch_bounds__(256)
void gru_gates_kernel(const float* __restrict__ gi,
                      const float* __restrict__ gh,
                      float* __restrict__ h,
                      float* __restrict__ save_f32,
                      _Float16* __restrict__ save_h16_base, int t_col, int nTiles)
{
    int i = blockIdx.x * 256 + threadIdx.x;
    if (i >= H_) return;
    float ir = gi[i], iz = gi[H_ + i], inn = gi[2 * H_ + i];
    float hr = gh[i], hz = gh[H_ + i], hn  = gh[2 * H_ + i];
    float r  = 1.0f / (1.0f + expf(-(ir + hr)));
    float z  = 1.0f / (1.0f + expf(-(iz + hz)));
    float nn = tanhf(inn + r * hn);
    float h2 = (1.0f - z) * nn + z * h[i];
    h[i] = h2;
    if (save_f32) save_f32[i] = h2;
    if (save_h16_base) save_h16_base[bswz_idx(i, t_col, nTiles)] = (_Float16)h2;
}

// softmax over 400 attention scores; writes normalized weights to aw and to
// the attns output row.
__global__ __launch_bounds__(512)
void softmax400_kernel(const float* __restrict__ a,
                       float* __restrict__ aw,
                       float* __restrict__ attn_out)
{
    __shared__ float red[512];
    const int tid = threadIdx.x;
    float m = -3.402823e38f;
    for (int l = tid; l < L_; l += 512) m = fmaxf(m, a[l]);
    red[tid] = m; __syncthreads();
    for (int w = 256; w > 0; w >>= 1) {
        if (tid < w) red[tid] = fmaxf(red[tid], red[tid + w]);
        __syncthreads();
    }
    const float mx = red[0];
    __syncthreads();
    float s = 0.0f;
    for (int l = tid; l < L_; l += 512) s += expf(a[l] - mx);
    red[tid] = s; __syncthreads();
    for (int w = 256; w > 0; w >>= 1) {
        if (tid < w) red[tid] += red[tid + w];
        __syncthreads();
    }
    const float inv = 1.0f / red[0];
    for (int l = tid; l < L_; l += 512) {
        float v = expf(a[l] - mx) * inv;
        aw[l] = v;
        attn_out[l] = v;
    }
}

// in-place log-softmax over one row of V_ logits per block
__global__ __launch_bounds__(256)
void logsoftmax_kernel(float* __restrict__ out)
{
    __shared__ float red[256];
    float* row = out + (size_t)blockIdx.x * V_;
    const int tid = threadIdx.x;
    float m = -3.402823e38f;
    for (int v = tid; v < V_; v += 256) m = fmaxf(m, row[v]);
    red[tid] = m; __syncthreads();
    for (int w = 128; w > 0; w >>= 1) {
        if (tid < w) red[tid] = fmaxf(red[tid], red[tid + w]);
        __syncthreads();
    }
    const float mx = red[0];
    __syncthreads();
    float s = 0.0f;
    for (int v = tid; v < V_; v += 256) s += expf(row[v] - mx);
    red[tid] = s; __syncthreads();
    for (int w = 128; w > 0; w >>= 1) {
        if (tid < w) red[tid] += red[tid + w];
        __syncthreads();
    }
    const float lse = mx + logf(red[0]);
    __syncthreads();
    for (int v = tid; v < V_; v += 256) row[v] -= lse;
}

// gather embeddings into swizzled f16 B fragments; shift_sos!=0 prepends SOS
// and shifts tokens by one (decoder teacher forcing); pads n>=realN with 0.
__global__ __launch_bounds__(256)
void gather_embB_kernel(const int* __restrict__ toks, int shift_sos,
                        const float* __restrict__ emb,
                        _Float16* __restrict__ B,
                        int nTiles, int realN, int total)
{
    int idx = blockIdx.x * 256 + threadIdx.x;
    if (idx >= total) return;
    int p    = idx & 15;
    int lane = (idx >> 4) & 31;
    int frag = idx >> 9;               // kTile * nTiles + nTile
    int nT = frag % nTiles;
    int kT = frag / nTiles;
    int k = kT * 32 + (lane >> 4) * 16 + p;
    int n = nT * 16 + (lane & 15);
    _Float16 v = (_Float16)0.0f;
    if (n < realN) {
        int tok = shift_sos ? ((n == 0) ? SOS_ : toks[n - 1]) : toks[n];
        v = (_Float16)emb[(size_t)tok * E_ + k];
    }
    B[idx] = v;
}

// zero h0 and the swizzled Hb buffer (ws is poisoned; pad columns must be 0)
__global__ __launch_bounds__(256)
void init_state_kernel(float* __restrict__ h, _Float16* __restrict__ Hb)
{
    int i = blockIdx.x * 256 + threadIdx.x;
    if (i < H_) h[i] = 0.0f;
    if (i < H_ * TPAD) Hb[i] = (_Float16)0.0f;
}

// =====================================================================
extern "C" void kernel_launch(void* const* d_in, const int* in_sizes, int n_in,
                              void* d_out, int out_size, void* d_ws, size_t ws_size,
                              hipStream_t stream) {
    (void)in_sizes; (void)n_in; (void)out_size; (void)ws_size;

    const int*   input_tokens  = (const int*)  d_in[0];
    const int*   target_tokens = (const int*)  d_in[1];
    const float* embedding     = (const float*)d_in[2];
    const float* enc_Wih       = (const float*)d_in[3];
    const float* enc_Whh       = (const float*)d_in[4];
    const float* enc_bih       = (const float*)d_in[5];
    const float* enc_bhh       = (const float*)d_in[6];
    const float* attn_W        = (const float*)d_in[7];
    const float* attn_b        = (const float*)d_in[8];
    const float* comb_W        = (const float*)d_in[9];
    const float* comb_b        = (const float*)d_in[10];
    const float* dec_Wih       = (const float*)d_in[11];
    const float* dec_Whh       = (const float*)d_in[12];
    const float* dec_bih       = (const float*)d_in[13];
    const float* dec_bhh       = (const float*)d_in[14];
    const float* out_W         = (const float*)d_in[15];
    const float* out_b         = (const float*)d_in[16];

    float* out = (float*)d_out;                 // [T,V] log_probs ++ [T,L] attns
    float* attns_out = out + (size_t)T_ * V_;

    const int nTilesL = L_ / 16;    // 25
    const int nTilesT = TPAD / 16;  // 7
    const int nPairsL = (nTilesL + 1) / 2;  // 13
    const int nPairsT = (nTilesT + 1) / 2;  // 4

    // ---- workspace bump allocator (deterministic layout every call) ----
    size_t off = 0;
    auto alloc = [&](size_t bytes) -> void* {
        void* p = (char*)d_ws + off;
        off += (bytes + 255) & ~(size_t)255;
        return p;
    };
    _Float16* encEmbB = (_Float16*)alloc((size_t)E_ * L_    * 2); // swizzled frags
    _Float16* decEmbB = (_Float16*)alloc((size_t)E_ * TPAD  * 2); // swizzled frags
    _Float16* Hb      = (_Float16*)alloc((size_t)H_ * TPAD  * 2); // swizzled frags
    float* encGI   = (float*)alloc((size_t)3 * H_ * L_   * 4);    // [t][3H]
    float* attnX   = (float*)alloc((size_t)L_ * TPAD     * 4);    // [t][L]
    float* combX   = (float*)alloc((size_t)H_ * TPAD     * 4);    // [t][H]
    float* encOuts = (float*)alloc((size_t)L_ * H_       * 4);    // [l][H]
    float* h   = (float*)alloc(H_ * 4);
    float* gh  = (float*)alloc(3 * H_ * 4);
    float* gi  = (float*)alloc(3 * H_ * 4);
    float* sc  = (float*)alloc(L_ * 4);
    float* aw  = (float*)alloc(L_ * 4);
    float* ctx = (float*)alloc(H_ * 4);
    float* g   = (float*)alloc(H_ * 4);

    // ---- phase 0: gathers + init ----
    gather_embB_kernel<<<(E_ * L_ + 255) / 256, 256, 0, stream>>>(
        input_tokens, 0, embedding, encEmbB, nTilesL, L_, E_ * L_);
    gather_embB_kernel<<<(E_ * TPAD + 255) / 256, 256, 0, stream>>>(
        target_tokens, 1, embedding, decEmbB, nTilesT, T_, E_ * TPAD);
    init_state_kernel<<<(H_ * TPAD + 255) / 256, 256, 0, stream>>>(h, Hb);

    // ---- phase 1: batched input-side GEMMs (WMMA) ----
    // encGI[t][m] = enc_Wih @ enc_emb_t + enc_bih
    wmma_gemm_kernel<<<dim3(3 * H_ / 16, nPairsL), 32, 0, stream>>>(
        enc_Wih, E_, (const v16h*)encEmbB, nTilesL, enc_bih, encGI, 3 * H_, E_, L_);
    // attnX[t][l] = attn_W[:, :H] @ dec_emb_t + attn_b
    wmma_gemm_kernel<<<dim3(L_ / 16, nPairsT), 32, 0, stream>>>(
        attn_W, 2 * H_, (const v16h*)decEmbB, nTilesT, attn_b, attnX, L_, E_, TPAD);
    // combX[t][i] = comb_W[:, :H] @ dec_emb_t + comb_b
    wmma_gemm_kernel<<<dim3(H_ / 16, nPairsT), 32, 0, stream>>>(
        comb_W, 2 * H_, (const v16h*)decEmbB, nTilesT, comb_b, combX, H_, E_, TPAD);

    // ---- phase 2: sequential encoder ----
    for (int t = 0; t < L_; ++t) {
        matvec_kernel<<<3 * H_, 256, 0, stream>>>(enc_Whh, H_, h, H_, enc_bhh, nullptr, gh, 0);
        gru_gates_kernel<<<(H_ + 255) / 256, 256, 0, stream>>>(
            encGI + (size_t)t * 3 * H_, gh, h, encOuts + (size_t)t * H_, nullptr, 0, 0);
    }

    // ---- phase 3: sequential attention decoder ----
    for (int t = 0; t < T_; ++t) {
        // attention scores: attnX[t] + attn_W[:, H:] @ h
        matvec_kernel<<<L_, 256, 0, stream>>>(attn_W + H_, 2 * H_, h, H_,
                                              nullptr, attnX + (size_t)t * L_, sc, 0);
        softmax400_kernel<<<1, 512, 0, stream>>>(sc, aw, attns_out + (size_t)t * L_);
        ctx_kernel<<<(H_ + 255) / 256, 256, 0, stream>>>(encOuts, aw, ctx);
        // g = relu(combX[t] + comb_W[:, H:] @ ctx)
        matvec_kernel<<<H_, 256, 0, stream>>>(comb_W + H_, 2 * H_, ctx, H_,
                                              nullptr, combX + (size_t)t * H_, g, 1);
        matvec_kernel<<<3 * H_, 256, 0, stream>>>(dec_Wih, H_, g, H_, dec_bih, nullptr, gi, 0);
        matvec_kernel<<<3 * H_, 256, 0, stream>>>(dec_Whh, H_, h, H_, dec_bhh, nullptr, gh, 0);
        gru_gates_kernel<<<(H_ + 255) / 256, 256, 0, stream>>>(
            gi, gh, h, nullptr, Hb, t, nTilesT);
    }

    // ---- phase 4: batched output projection (WMMA) + log-softmax ----
    // out[t][v] = out_W @ h2_t + out_b   (stores guarded to n < T)
    wmma_gemm_kernel<<<dim3(V_ / 16, nPairsT), 32, 0, stream>>>(
        out_W, H_, (const v16h*)Hb, nTilesT, out_b, out, V_, H_, T_);
    logsoftmax_kernel<<<T_, 256, 0, stream>>>(out);
}
